// LSTM_41420664603057
// MI455X (gfx1250) — compile-verified, matrix operand order checked by reading, not run on hardware
//
#include <hip/hip_runtime.h>
#include <hip/hip_bf16.h>

// ---- WMMA fragment types (gfx1250, wave32) ----
typedef __attribute__((ext_vector_type(16))) __bf16 v16bf;
typedef __attribute__((ext_vector_type(8)))  __bf16 v8bf;
typedef __attribute__((ext_vector_type(4)))  __bf16 v4bf;
typedef __attribute__((ext_vector_type(8)))  float  v8f;

#define M_TOT 1024
#define N_TOT 2048
#define K_X   1024
#define K_H   2048

#define BM 64
#define BN 64
#define BK 32
#define KSTEPS_X (K_X / BK)   // 32 x-tiles
#define KSTEPS_H (K_H / BK)   // 64 h-tiles
#define PITCH 40              // LDS row pitch in bf16 (32 data + 8 pad) -> conflict-free frag reads

// Branch-free fast activations (straight-line VALU/TRANS, no exec-mask divergence)
__device__ __forceinline__ float fsigmoid(float v) {
    return __builtin_amdgcn_rcpf(1.0f + __expf(-v));
}
__device__ __forceinline__ float ftanh(float v) {
    const float e = __expf(-2.0f * __builtin_fabsf(v));
    const float t = (1.0f - e) * __builtin_amdgcn_rcpf(1.0f + e);
    return __builtin_copysignf(t, v);
}

struct Stage {              // one K-step of staging data, pre-converted to bf16 (20 VGPRs)
    v4bf a[2];              // A tile: 2 x 4 elems per thread
    v4bf b[4][2];           // B tiles: 4 gates x 2 x 4 elems per thread
};

__global__ __launch_bounds__(256) void lstm_wmma_bf16(
    const float* __restrict__ x, const float* __restrict__ h,
    const float* __restrict__ c_old,
    const float* __restrict__ wf, const float* __restrict__ uf, const float* __restrict__ bfp,
    const float* __restrict__ wi, const float* __restrict__ ui, const float* __restrict__ bip,
    const float* __restrict__ wo, const float* __restrict__ uo, const float* __restrict__ bop,
    const float* __restrict__ wc, const float* __restrict__ uc, const float* __restrict__ bcp,
    float* __restrict__ out)
{
    __shared__ __attribute__((aligned(16))) __bf16 ldsA[BM * PITCH];        // [m][k]
    __shared__ __attribute__((aligned(16))) __bf16 ldsB[4][BN * PITCH];     // [gate][n][k] (transposed)

    const float* W[4] = {wf, wi, wo, wc};
    const float* U[4] = {uf, ui, uo, uc};

    const int tid  = threadIdx.x;
    const int lane = tid & 31;
    const int wid  = tid >> 5;
    const int wm   = wid >> 2;          // 0..1  (M direction)
    const int wn   = wid & 3;           // 0..3  (N direction)

    const int m0 = blockIdx.y * BM;
    const int n0 = blockIdx.x * BN;

    // Prefetch this block's c_old tile (64x64 fp32) for the epilogue -> global_prefetch_b8
    {
        const int row = tid >> 2;
        const int seg = tid & 3;
        __builtin_prefetch(&c_old[(size_t)(m0 + row) * N_TOT + n0 + seg * 16], 0, 0);
    }

    v8f acc[4][2];
    #pragma unroll
    for (int g = 0; g < 4; ++g)
        #pragma unroll
        for (int t = 0; t < 2; ++t)
            #pragma unroll
            for (int e = 0; e < 8; ++e) acc[g][t][e] = 0.0f;

    const int half = lane >> 4;         // 0/1
    const int lr   = lane & 15;
    const int cb   = half * 8;          // per-lane K chunk base within the tile

    // ---- pipeline helpers (source bases are loop-phase constants -> no per-iter selects) ----
    auto load_x = [&](Stage& s, int kb) {
        const int k0 = kb * BK;
        #pragma unroll
        for (int i = 0; i < 2; ++i) {
            const int idx = tid + i * 256;      // 0..511
            const int row = idx >> 3;           // 0..63
            const int c4  = idx & 7;
            const float4 v = *(const float4*)(x + (size_t)(m0 + row) * K_X + k0 + c4 * 4);
            s.a[i][0] = (__bf16)v.x; s.a[i][1] = (__bf16)v.y;
            s.a[i][2] = (__bf16)v.z; s.a[i][3] = (__bf16)v.w;
        }
        #pragma unroll
        for (int g = 0; g < 4; ++g)
            #pragma unroll
            for (int i = 0; i < 2; ++i) {
                const int idx = tid + i * 256;
                const int kr  = idx >> 4;       // 0..31
                const int c4  = idx & 15;
                const float4 v = *(const float4*)(W[g] + (size_t)(k0 + kr) * N_TOT + n0 + c4 * 4);
                s.b[g][i][0] = (__bf16)v.x; s.b[g][i][1] = (__bf16)v.y;
                s.b[g][i][2] = (__bf16)v.z; s.b[g][i][3] = (__bf16)v.w;
            }
    };

    auto load_h = [&](Stage& s, int kb) {       // kb relative to h-phase start
        const int k0 = kb * BK;
        #pragma unroll
        for (int i = 0; i < 2; ++i) {
            const int idx = tid + i * 256;
            const int row = idx >> 3;
            const int c4  = idx & 7;
            const float4 v = *(const float4*)(h + (size_t)(m0 + row) * K_H + k0 + c4 * 4);
            s.a[i][0] = (__bf16)v.x; s.a[i][1] = (__bf16)v.y;
            s.a[i][2] = (__bf16)v.z; s.a[i][3] = (__bf16)v.w;
        }
        #pragma unroll
        for (int g = 0; g < 4; ++g)
            #pragma unroll
            for (int i = 0; i < 2; ++i) {
                const int idx = tid + i * 256;
                const int kr  = idx >> 4;
                const int c4  = idx & 15;
                const float4 v = *(const float4*)(U[g] + (size_t)(k0 + kr) * N_TOT + n0 + c4 * 4);
                s.b[g][i][0] = (__bf16)v.x; s.b[g][i][1] = (__bf16)v.y;
                s.b[g][i][2] = (__bf16)v.z; s.b[g][i][3] = (__bf16)v.w;
            }
    };

    auto store_stage = [&](const Stage& s) {
        #pragma unroll
        for (int i = 0; i < 2; ++i) {
            const int idx = tid + i * 256;
            const int row = idx >> 3;
            const int c4  = idx & 7;
            *(v4bf*)(&ldsA[row * PITCH + c4 * 4]) = s.a[i];   // 8B aligned LDS store
        }
        #pragma unroll
        for (int g = 0; g < 4; ++g)
            #pragma unroll
            for (int i = 0; i < 2; ++i) {
                const int idx = tid + i * 256;
                const int kr  = idx >> 4;
                const int nl  = (idx & 15) * 4;
                ldsB[g][(nl + 0) * PITCH + kr] = s.b[g][i][0];  // transpose scatter
                ldsB[g][(nl + 1) * PITCH + kr] = s.b[g][i][1];
                ldsB[g][(nl + 2) * PITCH + kr] = s.b[g][i][2];
                ldsB[g][(nl + 3) * PITCH + kr] = s.b[g][i][3];
            }
    };

    auto compute = [&]() {
        v16bf aF[2];
        v16bf bF[4];
        #pragma unroll
        for (int t = 0; t < 2; ++t) {
            const __bf16* pa = &ldsA[(wm * 32 + t * 16 + lr) * PITCH + cb];
            const v8bf lo = *(const v8bf*)pa;
            const v8bf hi = *(const v8bf*)(pa + 16);
            #pragma unroll
            for (int e = 0; e < 8; ++e) { aF[t][e] = lo[e]; aF[t][e + 8] = hi[e]; }
        }
        #pragma unroll
        for (int g = 0; g < 4; ++g) {
            const __bf16* pb = &ldsB[g][(wn * 16 + lr) * PITCH + cb];
            const v8bf lo = *(const v8bf*)pb;
            const v8bf hi = *(const v8bf*)(pb + 16);
            #pragma unroll
            for (int e = 0; e < 8; ++e) { bF[g][e] = lo[e]; bF[g][e + 8] = hi[e]; }
        }
        #pragma unroll
        for (int g = 0; g < 4; ++g)
            #pragma unroll
            for (int t = 0; t < 2; ++t)
                acc[g][t] = __builtin_amdgcn_wmma_f32_16x16x32_bf16(
                    false, aF[t], false, bF[g], (short)0, acc[g][t], false, false);
    };

    // ---- software-pipelined main loops (phase-split: no per-iteration source selects) ----
    Stage s0, s1;

    // X phase: K tiles 0..31 over x/W
    load_x(s0, 0);
    for (int kb = 0; kb < KSTEPS_X; kb += 2) {
        __syncthreads();                 // previous compute done reading LDS
        store_stage(s0);
        load_x(s1, kb + 1);              // in flight during compute below
        __syncthreads();                 // LDS tile visible
        compute();

        __syncthreads();
        store_stage(s1);
        if (kb + 2 < KSTEPS_X) load_x(s0, kb + 2);
        else                   load_h(s0, 0);        // hand off pipeline into H phase
        __syncthreads();
        compute();
    }

    // H phase: K tiles 0..63 over h/U (s0 already holds h-tile 0)
    for (int kb = 0; kb < KSTEPS_H; kb += 2) {
        __syncthreads();
        store_stage(s0);
        load_h(s1, kb + 1);
        __syncthreads();
        compute();

        __syncthreads();
        store_stage(s1);
        if (kb + 2 < KSTEPS_H) load_h(s0, kb + 2);
        __syncthreads();
        compute();
    }

    // ---- fused LSTM epilogue (branch-free fast activations) ----
    const int ng = n0 + wn * 16 + lr;                 // global N (column)
    const float biasf = bfp[ng];
    const float biasi = bip[ng];
    const float biaso = bop[ng];
    const float biasc = bcp[ng];

    #pragma unroll
    for (int t = 0; t < 2; ++t) {
        #pragma unroll
        for (int r = 0; r < 8; ++r) {
            const int mg = m0 + wm * 32 + t * 16 + half * 8 + r;   // C/D layout: M = half*8 + r
            const float fg = fsigmoid(acc[0][t][r] + biasf);
            const float ig = fsigmoid(acc[1][t][r] + biasi);
            const float og = fsigmoid(acc[2][t][r] + biaso);
            const float cg = ftanh(acc[3][t][r] + biasc);
            const float cn = fg * c_old[(size_t)mg * N_TOT + ng] + ig * cg;
            const float hn = og * ftanh(cn);
            out[(size_t)mg * N_TOT + ng] = hn;                                   // h_new
            out[(size_t)M_TOT * N_TOT + (size_t)mg * N_TOT + ng] = cn;           // c_new
        }
    }
}

extern "C" void kernel_launch(void* const* d_in, const int* in_sizes, int n_in,
                              void* d_out, int out_size, void* d_ws, size_t ws_size,
                              hipStream_t stream) {
    // setup_inputs() dict order: x, h, c_old, then per gate g in [f,i,o,c]: w_g, u_g, b_g
    const float* x     = (const float*)d_in[0];
    const float* h     = (const float*)d_in[1];
    const float* c_old = (const float*)d_in[2];
    const float* wf = (const float*)d_in[3];
    const float* uf = (const float*)d_in[4];
    const float* bf = (const float*)d_in[5];
    const float* wi = (const float*)d_in[6];
    const float* ui = (const float*)d_in[7];
    const float* bi = (const float*)d_in[8];
    const float* wo = (const float*)d_in[9];
    const float* uo = (const float*)d_in[10];
    const float* bo = (const float*)d_in[11];
    const float* wc = (const float*)d_in[12];
    const float* uc = (const float*)d_in[13];
    const float* bc = (const float*)d_in[14];

    dim3 grid(N_TOT / BN, M_TOT / BM);   // 32 x 16 = 512 workgroups
    lstm_wmma_bf16<<<grid, 256, 0, stream>>>(
        x, h, c_old, wf, uf, bf, wi, ui, bi, wo, uo, bo, wc, uc, bc, (float*)d_out);
}
